// ConditionedSparseAttention_18846316494867
// MI455X (gfx1250) — compile-verified, matrix operand order checked by reading, not run on hardware
//
#include <hip/hip_runtime.h>
#include <hip/hip_bf16.h>

typedef unsigned short u16;
typedef unsigned int   u32;

#define B_      2
#define T_IN_   2048
#define T_COND_ 1024
#define D_      1024
#define H_      16
#define HD_     64
#define S_      (T_IN_ + T_COND_)   /* 3072 */
#define W_      512
#define KW_     1024                /* window keys = 2*W */
#define MTOT_   (B_ * S_)           /* 6144 */

typedef __attribute__((ext_vector_type(16))) __bf16 v16bf;
typedef __attribute__((ext_vector_type(8)))  float  v8f;

union FragBF { v16bf v; u32 u[8]; };

__device__ __forceinline__ u16 f2bf(float f) {
  union { float f; u32 u; } x; x.f = f;
  u32 u = x.u;
  u += 0x7FFFu + ((u >> 16) & 1u);   // round-to-nearest-even
  return (u16)(u >> 16);
}

// A/B bf16 fragment: lane (l&15) = row/col index; VGPR r holds the K pair at:
__device__ __forceinline__ int frag_k(int lane, int r) {
  return ((lane >> 4) << 3) + ((r & 3) << 1) + ((r >> 2) << 4);
}

// Load a 16x32 bf16 fragment: 8 aligned dword (bf16-pair) loads.
__device__ __forceinline__ void load_frag(FragBF& f, const u16* base, int lane) {
#pragma unroll
  for (int r = 0; r < 8; ++r)
    f.u[r] = *(const u32*)(base + frag_k(lane, r));
}

// ---------------------------------------------------------------------------
// f32 -> bf16 staging: 8 elems/thread (2x float4 load, 1x b128 store)
// ---------------------------------------------------------------------------
__global__ void cvt_f32_bf16(const float* __restrict__ src,
                             u16* __restrict__ dst, int n8) {
  int i = blockIdx.x * blockDim.x + threadIdx.x;
  const int stride = gridDim.x * blockDim.x;
  for (; i < n8; i += stride) {
    const float4* s = (const float4*)src + 2 * (size_t)i;
    const float4 f0 = s[0], f1 = s[1];
    union { u16 h[8]; uint4 v; } o;
    o.h[0] = f2bf(f0.x); o.h[1] = f2bf(f0.y); o.h[2] = f2bf(f0.z); o.h[3] = f2bf(f0.w);
    o.h[4] = f2bf(f1.x); o.h[5] = f2bf(f1.y); o.h[6] = f2bf(f1.z); o.h[7] = f2bf(f1.w);
    ((uint4*)dst)[i] = o.v;
  }
}

// ---------------------------------------------------------------------------
// QKV projection: qkv[m,n] = sum_k inp[m,k] * Wqkv[n,k] + b[n]
// Two-stage software-pipelined WMMA main loop (k step 64 = 2x32).
// Epilogue: Q (scaled 1/8) -> q[b,h,s,hd]; K/V gathered into 1024-key window
// buffers (V stored transposed along keys for aligned PV fragment loads).
// Block = 8 waves (2 row x 4 col), wave tile 16x64, block tile 32x256.
// ---------------------------------------------------------------------------
__global__ __launch_bounds__(256)
void qkv_gemm(const u16* __restrict__ inp, const u16* __restrict__ w,
              const float* __restrict__ bias, const int* __restrict__ end_inds,
              u16* __restrict__ q, u16* __restrict__ kwin, u16* __restrict__ vtwin) {
  const int lane = threadIdx.x & 31;
  const int wave = threadIdx.x >> 5;
  const int bm = blockIdx.x % (MTOT_ / 32);
  const int bn = blockIdx.x / (MTOT_ / 32);
  const int m0 = bm * 32 + (wave & 1) * 16;
  const int n0 = bn * 256 + (wave >> 1) * 64;

  const u16* arow = inp + (size_t)(m0 + (lane & 15)) * D_;
  const u16* brow[4];
#pragma unroll
  for (int t = 0; t < 4; ++t)
    brow[t] = w + (size_t)(n0 + t * 16 + (lane & 15)) * D_;

  v8f acc[4] = {};
  FragBF aA, aB, bA[4], bB[4];

  load_frag(aA, arow, lane);
#pragma unroll
  for (int t = 0; t < 4; ++t) load_frag(bA[t], brow[t], lane);

  for (int k0 = 0; k0 < D_; k0 += 64) {
    // issue stage-B loads (k0+32) before consuming stage A
    load_frag(aB, arow + k0 + 32, lane);
#pragma unroll
    for (int t = 0; t < 4; ++t) load_frag(bB[t], brow[t] + k0 + 32, lane);
    __builtin_prefetch(arow + k0 + 128, 0, 1);
#pragma unroll
    for (int t = 0; t < 4; ++t)
      acc[t] = __builtin_amdgcn_wmma_f32_16x16x32_bf16(
          false, aA.v, false, bA[t].v, (short)0, acc[t], false, false);
    if (k0 + 64 < D_) {
      load_frag(aA, arow + k0 + 64, lane);
#pragma unroll
      for (int t = 0; t < 4; ++t) load_frag(bA[t], brow[t] + k0 + 64, lane);
    }
#pragma unroll
    for (int t = 0; t < 4; ++t)
      acc[t] = __builtin_amdgcn_wmma_f32_16x16x32_bf16(
          false, aB.v, false, bB[t].v, (short)0, acc[t], false, false);
  }

  const int bb = m0 / S_;            // batch (constant per wave: 32 | 3072)
  const int e  = end_inds[bb];
#pragma unroll
  for (int t = 0; t < 4; ++t) {
    const int n = n0 + t * 16 + (lane & 15);
    const float bv = bias[n];
    const int seg = n >> 10;         // 0=Q 1=K 2=V
    const int f = n & (D_ - 1);
    const int h = f >> 6, d = f & 63;
#pragma unroll
    for (int r = 0; r < 8; ++r) {
      const int m = m0 + r + ((lane >> 4) << 3);
      const int s = m - bb * S_;
      const float v = acc[t][r] + bv;
      if (seg == 0) {
        q[((size_t)(bb * H_ + h) * S_ + s) * HD_ + d] = f2bf(v * 0.125f);
      } else {
        int j = -1;
        if (s >= e - W_ && s < e)                      j = s - (e - W_);
        else if (s >= T_IN_ + e - W_ && s < T_IN_ + e) j = W_ + (s - T_IN_) - (e - W_);
        if (j >= 0) {
          if (seg == 1)
            kwin[((size_t)(bb * H_ + h) * KW_ + j) * HD_ + d] = f2bf(v);
          else
            vtwin[((size_t)(bb * H_ + h) * HD_ + d) * KW_ + j] = f2bf(v);
        }
      }
    }
  }
}

// ---------------------------------------------------------------------------
// Windowed flash attention: 1 wave per (b,h,16-query tile), 1024 window keys
// in 32-key chunks. Scores: 4 WMMAs; V fragments loaded early so their
// latency hides behind softmax; P re-layout via LDS; PV: 4 WMMAs.
// ---------------------------------------------------------------------------
__global__ __launch_bounds__(32)
void attn_win(const u16* __restrict__ q, const u16* __restrict__ kwin,
              const u16* __restrict__ vtwin, u16* __restrict__ out) {
  __shared__ u16 pl[16 * 32];
  const int lane = threadIdx.x;
  int id = blockIdx.x;
  const int qt = id % (S_ / 16); id /= (S_ / 16);
  const int h  = id % H_;        id /= H_;
  const int b  = id;
  const size_t bh = (size_t)(b * H_ + h);

  // Q A-fragments (d=0..31 and d=32..63), 1/sqrt(HD) already folded in
  FragBF qa0, qa1;
  const u16* qp = q + (bh * S_ + (size_t)qt * 16 + (lane & 15)) * HD_;
  load_frag(qa0, qp, lane);
  load_frag(qa1, qp + 32, lane);

  v8f acc[4] = {};
  float mrun[8], lsum[8];
#pragma unroll
  for (int r = 0; r < 8; ++r) { mrun[r] = -1e30f; lsum[r] = 0.f; }

  const u16* kbase = kwin  + bh * KW_ * HD_;
  const u16* vbase = vtwin + bh * HD_ * KW_;

  for (int j0 = 0; j0 < KW_; j0 += 32) {
    // ---- scores: 16 queries x 32 keys (two 16-key C tiles) ----
    v8f s0 = {}, s1 = {};
    {
      const u16* kp0 = kbase + (size_t)(j0 + (lane & 15)) * HD_;
      const u16* kp1 = kp0 + 16 * HD_;
      FragBF b00, b01, b10, b11;
      load_frag(b00, kp0,      lane);
      load_frag(b01, kp0 + 32, lane);
      load_frag(b10, kp1,      lane);
      load_frag(b11, kp1 + 32, lane);
      __builtin_prefetch(kp0 + 32 * HD_, 0, 1);   // next K chunk
      s0 = __builtin_amdgcn_wmma_f32_16x16x32_bf16(false, qa0.v, false, b00.v, (short)0, s0, false, false);
      s0 = __builtin_amdgcn_wmma_f32_16x16x32_bf16(false, qa1.v, false, b01.v, (short)0, s0, false, false);
      s1 = __builtin_amdgcn_wmma_f32_16x16x32_bf16(false, qa0.v, false, b10.v, (short)0, s1, false, false);
      s1 = __builtin_amdgcn_wmma_f32_16x16x32_bf16(false, qa1.v, false, b11.v, (short)0, s1, false, false);
    }

    // ---- issue V fragment loads now; consumed after softmax ----
    FragBF vb[4];
#pragma unroll
    for (int t4 = 0; t4 < 4; ++t4)
      load_frag(vb[t4], vbase + (size_t)(t4 * 16 + (lane & 15)) * KW_ + j0, lane);

    // ---- online softmax: row reductions across the 16-lane halves ----
    float p0[8], p1[8], corr[8];
#pragma unroll
    for (int r = 0; r < 8; ++r) {
      float t = fmaxf(s0[r], s1[r]);
      t = fmaxf(t, __shfl_xor(t, 1, 32));
      t = fmaxf(t, __shfl_xor(t, 2, 32));
      t = fmaxf(t, __shfl_xor(t, 4, 32));
      t = fmaxf(t, __shfl_xor(t, 8, 32));
      const float nm = fmaxf(mrun[r], t);
      corr[r] = __expf(mrun[r] - nm);
      p0[r] = __expf(s0[r] - nm);
      p1[r] = __expf(s1[r] - nm);
      float rs = p0[r] + p1[r];
      rs += __shfl_xor(rs, 1, 32);
      rs += __shfl_xor(rs, 2, 32);
      rs += __shfl_xor(rs, 4, 32);
      rs += __shfl_xor(rs, 8, 32);
      lsum[r] = lsum[r] * corr[r] + rs;
      mrun[r] = nm;
    }
#pragma unroll
    for (int t4 = 0; t4 < 4; ++t4)
#pragma unroll
      for (int r = 0; r < 8; ++r) acc[t4][r] *= corr[r];

    // ---- P: C-layout -> bf16 row-major in LDS -> A-layout fragment ----
    {
      const int n  = lane & 15;
      const int mb = (lane >> 4) << 3;
#pragma unroll
      for (int r = 0; r < 8; ++r) {
        pl[(mb + r) * 32 + n]      = f2bf(p0[r]);
        pl[(mb + r) * 32 + 16 + n] = f2bf(p1[r]);
      }
    }
    __syncthreads();
    FragBF pA;
    {
      const int m = lane & 15;
#pragma unroll
      for (int r = 0; r < 8; ++r)
        pA.u[r] = *(const u32*)&pl[m * 32 + frag_k(lane, r)];
    }
    __syncthreads();

    // ---- P (16x32) @ V (32x64): 4 WMMAs ----
#pragma unroll
    for (int t4 = 0; t4 < 4; ++t4)
      acc[t4] = __builtin_amdgcn_wmma_f32_16x16x32_bf16(
          false, pA.v, false, vb[t4].v, (short)0, acc[t4], false, false);
  }

  // ---- normalize + store to (B,S,D) bf16 for out-projection ----
#pragma unroll
  for (int t4 = 0; t4 < 4; ++t4) {
    const int n = lane & 15;
#pragma unroll
    for (int r = 0; r < 8; ++r) {
      const int m = r + ((lane >> 4) << 3);
      const float v = acc[t4][r] / lsum[r];
      out[((size_t)b * S_ + qt * 16 + m) * D_ + h * 64 + t4 * 16 + n] = f2bf(v);
    }
  }
}

// ---------------------------------------------------------------------------
// Out projection: out[m,n] = sum_k att[m,k] * Wout[n,k] + bout[n]  (f32 out)
// Same two-stage pipelined WMMA loop.
// ---------------------------------------------------------------------------
__global__ __launch_bounds__(256)
void out_gemm(const u16* __restrict__ a, const u16* __restrict__ w,
              const float* __restrict__ bias, float* __restrict__ out) {
  const int lane = threadIdx.x & 31;
  const int wave = threadIdx.x >> 5;
  const int bm = blockIdx.x % (MTOT_ / 32);
  const int bn = blockIdx.x / (MTOT_ / 32);
  const int m0 = bm * 32 + (wave & 1) * 16;
  const int n0 = bn * 256 + (wave >> 1) * 64;

  const u16* arow = a + (size_t)(m0 + (lane & 15)) * D_;
  const u16* brow[4];
#pragma unroll
  for (int t = 0; t < 4; ++t)
    brow[t] = w + (size_t)(n0 + t * 16 + (lane & 15)) * D_;

  v8f acc[4] = {};
  FragBF aA, aB, bA[4], bB[4];

  load_frag(aA, arow, lane);
#pragma unroll
  for (int t = 0; t < 4; ++t) load_frag(bA[t], brow[t], lane);

  for (int k0 = 0; k0 < D_; k0 += 64) {
    load_frag(aB, arow + k0 + 32, lane);
#pragma unroll
    for (int t = 0; t < 4; ++t) load_frag(bB[t], brow[t] + k0 + 32, lane);
    __builtin_prefetch(arow + k0 + 128, 0, 1);
#pragma unroll
    for (int t = 0; t < 4; ++t)
      acc[t] = __builtin_amdgcn_wmma_f32_16x16x32_bf16(
          false, aA.v, false, bA[t].v, (short)0, acc[t], false, false);
    if (k0 + 64 < D_) {
      load_frag(aA, arow + k0 + 64, lane);
#pragma unroll
      for (int t = 0; t < 4; ++t) load_frag(bA[t], brow[t] + k0 + 64, lane);
    }
#pragma unroll
    for (int t = 0; t < 4; ++t)
      acc[t] = __builtin_amdgcn_wmma_f32_16x16x32_bf16(
          false, aB.v, false, bB[t].v, (short)0, acc[t], false, false);
  }

#pragma unroll
  for (int t = 0; t < 4; ++t) {
    const int n = n0 + t * 16 + (lane & 15);
    const float bv = bias[n];
#pragma unroll
    for (int r = 0; r < 8; ++r) {
      const int m = m0 + r + ((lane >> 4) << 3);
      out[(size_t)m * D_ + n] = acc[t][r] + bv;
    }
  }
}

// ---------------------------------------------------------------------------
extern "C" void kernel_launch(void* const* d_in, const int* in_sizes, int n_in,
                              void* d_out, int out_size, void* d_ws, size_t ws_size,
                              hipStream_t stream) {
  const float* x    = (const float*)d_in[0];
  const float* cond = (const float*)d_in[1];
  const int*   ei   = (const int*)d_in[2];
  const float* wqkv = (const float*)d_in[3];
  const float* bqkv = (const float*)d_in[4];
  const float* wout = (const float*)d_in[5];
  const float* bout = (const float*)d_in[6];
  float* out = (float*)d_out;

  char* ws = (char*)d_ws;
  u16* inp = (u16*)ws; ws += (size_t)MTOT_ * D_ * 2;            // 12.6 MB
  u16* wq  = (u16*)ws; ws += (size_t)3 * D_ * D_ * 2;           //  6.3 MB
  u16* wo  = (u16*)ws; ws += (size_t)D_ * D_ * 2;               //  2.1 MB
  u16* qb  = (u16*)ws; ws += (size_t)MTOT_ * D_ * 2;            // 12.6 MB
  u16* kw  = (u16*)ws; ws += (size_t)B_ * H_ * KW_ * HD_ * 2;   //  4.2 MB
  u16* vtw = (u16*)ws; ws += (size_t)B_ * H_ * HD_ * KW_ * 2;   //  4.2 MB
  u16* att = (u16*)ws; ws += (size_t)MTOT_ * D_ * 2;            // 12.6 MB

  // 1) f32 -> bf16 staging (concat x|cond per batch, plus both weight mats)
  for (int b = 0; b < B_; ++b) {
    cvt_f32_bf16<<<256, 256, 0, stream>>>(
        x + (size_t)b * T_IN_ * D_, inp + (size_t)b * S_ * D_, T_IN_ * D_ / 8);
    cvt_f32_bf16<<<256, 256, 0, stream>>>(
        cond + (size_t)b * T_COND_ * D_,
        inp + (size_t)b * S_ * D_ + (size_t)T_IN_ * D_, T_COND_ * D_ / 8);
  }
  cvt_f32_bf16<<<256, 256, 0, stream>>>(wqkv, wq, 3 * D_ * D_ / 8);
  cvt_f32_bf16<<<256, 256, 0, stream>>>(wout, wo, D_ * D_ / 8);

  // 2) QKV projection + window gather
  qkv_gemm<<<(MTOT_ / 32) * (3 * D_ / 256), 256, 0, stream>>>(
      inp, wq, bqkv, ei, qb, kw, vtw);

  // 3) windowed flash attention (mask is query-independent: 1024 keys/batch)
  attn_win<<<B_ * H_ * (S_ / 16), 32, 0, stream>>>(qb, kw, vtw, att);

  // 4) output projection
  out_gemm<<<(MTOT_ / 32) * (D_ / 256), 256, 0, stream>>>(att, wo, bout, out);

  (void)in_sizes; (void)n_in; (void)out_size; (void)ws_size;
}
